// TripletInteraction_14714557956333
// MI455X (gfx1250) — matrix operand order repaired and checked by reading, not compiled, not run on hardware
//
#include <hip/hip_runtime.h>
#include <cmath>

#define E_CNT  250000
#define NBR    8
#define D_EDGE 128
#define D_RBF  16
#define D_CBF  16
#define D_TRI  64

typedef __attribute__((ext_vector_type(16))) __bf16 v16bf;
typedef __attribute__((ext_vector_type(8)))  __bf16 v8bf;
typedef __attribute__((ext_vector_type(8)))  float  v8f;

#define WMMA_BF16(a, b, c) \
  __builtin_amdgcn_wmma_f32_16x16x32_bf16(false, (a), false, (b), (short)0, (c), false, false)

// silu via hardware exp + hardware reciprocal (v_exp_f32 + v_rcp_f32):
// matches bf16-pipeline accuracy while avoiding the IEEE division expansion.
__device__ __forceinline__ float silu_f(float v) {
  return v * __builtin_amdgcn_rcpf(1.0f + __expf(-v));
}

// ---- fragment loaders -------------------------------------------------------
// A-matrix 16x32 bf16 layout (wave32): lanes 0-15 -> M=0..15, K = kb+{0..7,16..23}
//                                      lanes 16-31 -> M=0..15, K = kb+{8..15,24..31}

// A fragment from an f32 row-major source (converted on load).
__device__ __forceinline__ v16bf load_a32f(const float* __restrict__ src, int ld,
                                           int row0, int kb, int lane) {
  int m  = lane & 15;
  int k0 = kb + ((lane & 16) ? 8 : 0);
  const float* p = src + (size_t)(row0 + m) * ld + k0;
  v16bf a;
#pragma unroll
  for (int i = 0; i < 8; ++i) {
    a[i]     = (__bf16)p[i];
    a[8 + i] = (__bf16)p[16 + i];
  }
  return a;
}

// A fragment from f32 source with K=16, zero-padded to K=32.
__device__ __forceinline__ v16bf load_a16f(const float* __restrict__ src, int ld,
                                           int row0, int lane) {
  int m  = lane & 15;
  int k0 = (lane & 16) ? 8 : 0;
  const float* p = src + (size_t)(row0 + m) * ld + k0;
  v16bf a;
#pragma unroll
  for (int i = 0; i < 8; ++i) {
    a[i]     = (__bf16)p[i];
    a[8 + i] = (__bf16)0.0f;   // K 16..31 padded (B side is zero-padded too)
  }
  return a;
}

// A fragment from a bf16 row-major source (pure vector loads, no cvt).
__device__ __forceinline__ v16bf load_a_bf(const __bf16* __restrict__ src, int ld,
                                           int row0, int kb, int lane) {
  int m  = lane & 15;
  int k0 = kb + ((lane & 16) ? 8 : 0);
  const __bf16* p = src + (size_t)(row0 + m) * ld + k0;
  v8bf lo = *(const v8bf*)p;
  v8bf hi = *(const v8bf*)(p + 16);
  v16bf a;
#pragma unroll
  for (int i = 0; i < 8; ++i) { a[i] = lo[i]; a[8 + i] = hi[i]; }
  return a;
}

// B-matrix 32x16 bf16 layout (wave32): lanes 0-15 -> N=n0..n0+15, K = kb+{0..7,16..23}
//                                      lanes 16-31 -> N=n0..n0+15, K = kb+{8..15,24..31}
// Weights are pre-converted bf16, row-major, K padded to a multiple of 32.
__device__ __forceinline__ v16bf load_b_bf(const __bf16* __restrict__ W, int ld,
                                           int kb, int n0, int lane) {
  int k0 = kb + ((lane & 16) ? 8 : 0);
  const __bf16* p = W + (size_t)k0 * ld + n0 + (lane & 15);
  v16bf b;
#pragma unroll
  for (int i = 0; i < 8; ++i) { b[i] = p[(size_t)i * ld]; b[8 + i] = p[(size_t)(16 + i) * ld]; }
  return b;
}

// ---- kernel 0: one-shot weight conversion f32 -> bf16 (with K padding) ------
__global__ void k0_prep(const float* __restrict__ Wmr, const float* __restrict__ Wr,
                        const float* __restrict__ Wmc, const float* __restrict__ Wc,
                        const float* __restrict__ Wd,  const float* __restrict__ Wst,
                        const float* __restrict__ Wts, __bf16* __restrict__ bW) {
  int t = blockIdx.x * blockDim.x + threadIdx.x;
  int stride = gridDim.x * blockDim.x;
  __bf16* bWmr = bW;                 // 128x128
  __bf16* bWr  = bWmr + 16384;       // 32x128 (K 16..31 zero)
  __bf16* bWmc = bWr  + 4096;        // 128x64
  __bf16* bWc  = bWmc + 8192;        // 32x64  (K 16..31 zero)
  __bf16* bWd  = bWc  + 2048;        // 64x128
  __bf16* bWst = bWd  + 8192;        // 128x128
  __bf16* bWts = bWst + 16384;       // 128x128
  for (int i = t; i < 16384; i += stride) bWmr[i] = (__bf16)Wmr[i];
  for (int i = t; i < 4096;  i += stride) {
    int k = i >> 7, n = i & 127;
    bWr[i] = (k < 16) ? (__bf16)Wr[k * 128 + n] : (__bf16)0.0f;
  }
  for (int i = t; i < 8192;  i += stride) bWmc[i] = (__bf16)Wmc[i];
  for (int i = t; i < 2048;  i += stride) {
    int k = i >> 6, n = i & 63;
    bWc[i] = (k < 16) ? (__bf16)Wc[k * 64 + n] : (__bf16)0.0f;
  }
  for (int i = t; i < 8192;  i += stride) bWd[i]  = (__bf16)Wd[i];
  for (int i = t; i < 16384; i += stride) bWst[i] = (__bf16)Wst[i];
  for (int i = t; i < 16384; i += stride) bWts[i] = (__bf16)Wts[i];
}

// ---- kernel 1: mnb = silu( (silu(m_st@Wmr) * (rbf@Wr)) @ Wmc ) --------------
__global__ __launch_bounds__(160) void k1_rbf_mcbf(
    const float* __restrict__ m_st, const float* __restrict__ rbf,
    const __bf16* __restrict__ bWmr, const __bf16* __restrict__ bWr,
    const __bf16* __restrict__ bWmc, float* __restrict__ mnb) {
  __shared__ __bf16 sT[5][16 * D_EDGE];
  int lane = threadIdx.x & 31, wv = threadIdx.x >> 5;
  int r0   = (blockIdx.x * 5 + wv) * 16;
  int colg = lane & 15, rowg = (lane >> 4) * 8;

  v16bf a[4];
#pragma unroll
  for (int kc = 0; kc < 4; ++kc) a[kc] = load_a32f(m_st, D_EDGE, r0, kc * 32, lane);
  v16bf a2 = load_a16f(rbf, D_RBF, r0, lane);

#pragma unroll
  for (int nt = 0; nt < 8; ++nt) {
    v8f acc = {};
#pragma unroll
    for (int kc = 0; kc < 4; ++kc)
      acc = WMMA_BF16(a[kc], load_b_bf(bWmr, D_EDGE, kc * 32, nt * 16, lane), acc);
    v8f accr = {};
    accr = WMMA_BF16(a2, load_b_bf(bWr, D_EDGE, 0, nt * 16, lane), accr);
#pragma unroll
    for (int r = 0; r < 8; ++r) {
      float v = silu_f(acc[r]) * accr[r];
      sT[wv][(rowg + r) * D_EDGE + nt * 16 + colg] = (__bf16)v;   // C-layout -> bf16 tile
    }
  }
  __syncthreads();

  v16bf am[4];
#pragma unroll
  for (int kc = 0; kc < 4; ++kc) am[kc] = load_a_bf(&sT[wv][0], D_EDGE, 0, kc * 32, lane);
#pragma unroll
  for (int nt = 0; nt < 4; ++nt) {
    v8f acc = {};
#pragma unroll
    for (int kc = 0; kc < 4; ++kc)
      acc = WMMA_BF16(am[kc], load_b_bf(bWmc, D_TRI, kc * 32, nt * 16, lane), acc);
#pragma unroll
    for (int r = 0; r < 8; ++r)
      mnb[(size_t)(r0 + rowg + r) * D_TRI + nt * 16 + colg] = silu_f(acc[r]);
  }
}

// ---- kernel 2: gather + cbf branch + W_dir, x stored as bf16 ----------------
__global__ __launch_bounds__(160) void k2_tri_dir(
    const float* __restrict__ cbf,  const int* __restrict__ idx_s,
    const int* __restrict__ basis,  const float* __restrict__ mnb,
    const __bf16* __restrict__ bWc, const __bf16* __restrict__ bWd,
    __bf16* __restrict__ xb) {
  __shared__ __bf16 sT[5][16 * D_TRI];
  int lane = threadIdx.x & 31, wv = threadIdx.x >> 5;
  int r0   = (blockIdx.x * 5 + wv) * 16;
  int colg = lane & 15, rowg = (lane >> 4) * 8;

  int is_ = idx_s[r0 + colg];   // this lane owns row (lane&15)
  v8f accx[4] = {{}, {}, {}, {}};

  for (int nb = 0; nb < NBR; ++nb) {
    int bidx = basis[(size_t)is_ * NBR + nb];
    int rowIdx[8];
#pragma unroll
    for (int r = 0; r < 8; ++r) rowIdx[r] = __shfl(bidx, rowg + r, 32);

    v16bf ac = load_a16f(cbf + nb * D_CBF, NBR * D_CBF, r0, lane);
#pragma unroll
    for (int nt = 0; nt < 4; ++nt) {
      v8f t = {};
      t = WMMA_BF16(ac, load_b_bf(bWc, D_TRI, 0, nt * 16, lane), t);
#pragma unroll
      for (int r = 0; r < 8; ++r) {
        float g = mnb[(size_t)rowIdx[r] * D_TRI + nt * 16 + colg];   // L2-resident gather
        accx[nt][r] += t[r] * g;
      }
    }
  }

  const float inv_sqrt_nb = 0.35355339059327373f;
#pragma unroll
  for (int nt = 0; nt < 4; ++nt)
#pragma unroll
    for (int r = 0; r < 8; ++r)
      sT[wv][(rowg + r) * D_TRI + nt * 16 + colg] = (__bf16)(accx[nt][r] * inv_sqrt_nb);
  __syncthreads();

  v16bf ax[2];
#pragma unroll
  for (int kc = 0; kc < 2; ++kc) ax[kc] = load_a_bf(&sT[wv][0], D_TRI, 0, kc * 32, lane);
#pragma unroll
  for (int nt = 0; nt < 8; ++nt) {
    v8f acc = {};
#pragma unroll
    for (int kc = 0; kc < 2; ++kc)
      acc = WMMA_BF16(ax[kc], load_b_bf(bWd, D_EDGE, kc * 32, nt * 16, lane), acc);
#pragma unroll
    for (int r = 0; r < 8; ++r)
      xb[(size_t)(r0 + rowg + r) * D_EDGE + nt * 16 + colg] = (__bf16)silu_f(acc[r]);
  }
}

// ---- kernel 3: xts = silu(x @ W_ts) ----------------------------------------
__global__ __launch_bounds__(160) void k3_ts(
    const __bf16* __restrict__ xb, const __bf16* __restrict__ bWts,
    float* __restrict__ xts) {
  int lane = threadIdx.x & 31, wv = threadIdx.x >> 5;
  int r0   = (blockIdx.x * 5 + wv) * 16;
  int colg = lane & 15, rowg = (lane >> 4) * 8;

  v16bf a[4];
#pragma unroll
  for (int kc = 0; kc < 4; ++kc) a[kc] = load_a_bf(xb, D_EDGE, r0, kc * 32, lane);
#pragma unroll
  for (int nt = 0; nt < 8; ++nt) {
    v8f acc = {};
#pragma unroll
    for (int kc = 0; kc < 4; ++kc)
      acc = WMMA_BF16(a[kc], load_b_bf(bWts, D_EDGE, kc * 32, nt * 16, lane), acc);
#pragma unroll
    for (int r = 0; r < 8; ++r)
      xts[(size_t)(r0 + rowg + r) * D_EDGE + nt * 16 + colg] = silu_f(acc[r]);
  }
}

// ---- kernel 4: out = (silu(x@W_st) + xts[idx_swap]) / sqrt(2) ---------------
__global__ __launch_bounds__(160) void k4_st_combine(
    const __bf16* __restrict__ xb,  const __bf16* __restrict__ bWst,
    const float* __restrict__ xts,  const int* __restrict__ idx_swap,
    float* __restrict__ out) {
  int lane = threadIdx.x & 31, wv = threadIdx.x >> 5;
  int r0   = (blockIdx.x * 5 + wv) * 16;
  int colg = lane & 15, rowg = (lane >> 4) * 8;
  const float inv_sqrt_2 = 0.7071067811865476f;

  int isw = idx_swap[r0 + colg];
  int rowIdx[8];
#pragma unroll
  for (int r = 0; r < 8; ++r) rowIdx[r] = __shfl(isw, rowg + r, 32);

  v16bf a[4];
#pragma unroll
  for (int kc = 0; kc < 4; ++kc) a[kc] = load_a_bf(xb, D_EDGE, r0, kc * 32, lane);
#pragma unroll
  for (int nt = 0; nt < 8; ++nt) {
    v8f acc = {};
#pragma unroll
    for (int kc = 0; kc < 4; ++kc)
      acc = WMMA_BF16(a[kc], load_b_bf(bWst, D_EDGE, kc * 32, nt * 16, lane), acc);
#pragma unroll
    for (int r = 0; r < 8; ++r) {
      float g = xts[(size_t)rowIdx[r] * D_EDGE + nt * 16 + colg];  // L2-resident gather
      out[(size_t)(r0 + rowg + r) * D_EDGE + nt * 16 + colg] =
          (silu_f(acc[r]) + g) * inv_sqrt_2;
    }
  }
}

extern "C" void kernel_launch(void* const* d_in, const int* in_sizes, int n_in,
                              void* d_out, int out_size, void* d_ws, size_t ws_size,
                              hipStream_t stream) {
  const float* m_st  = (const float*)d_in[0];
  const float* rbf   = (const float*)d_in[1];
  const float* cbf   = (const float*)d_in[2];
  const int*   idx_s = (const int*)  d_in[3];
  const int*   idxsw = (const int*)  d_in[4];
  const int*   basis = (const int*)  d_in[5];
  const float* Wmr   = (const float*)d_in[6];
  const float* Wr    = (const float*)d_in[7];
  const float* Wmc   = (const float*)d_in[8];
  const float* Wc    = (const float*)d_in[9];
  const float* Wd    = (const float*)d_in[10];
  const float* Wst   = (const float*)d_in[11];
  const float* Wts   = (const float*)d_in[12];

  // Workspace layout (all 32B-aligned):
  float*  ws   = (float*)d_ws;
  float*  mnb  = ws;                                      // E*64  f32
  float*  xts  = mnb + (size_t)E_CNT * D_TRI;             // E*128 f32
  __bf16* xb   = (__bf16*)(xts + (size_t)E_CNT * D_EDGE); // E*128 bf16
  __bf16* bW   = xb + (size_t)E_CNT * D_EDGE;             // 71680 bf16 weights
  __bf16* bWmr = bW;
  __bf16* bWr  = bWmr + 16384;
  __bf16* bWmc = bWr  + 4096;
  __bf16* bWc  = bWmc + 8192;
  __bf16* bWd  = bWc  + 2048;
  __bf16* bWst = bWd  + 8192;
  __bf16* bWts = bWst + 16384;

  dim3 blk(160);        // 5 waves/block, 1 tile of 16 edges per wave
  dim3 grd(3125);       // 3125 * 5 * 16 = 250000 = E exactly

  k0_prep       <<<64, 256, 0, stream>>>(Wmr, Wr, Wmc, Wc, Wd, Wst, Wts, bW);
  k1_rbf_mcbf   <<<grd, blk, 0, stream>>>(m_st, rbf, bWmr, bWr, bWmc, mnb);
  k2_tri_dir    <<<grd, blk, 0, stream>>>(cbf, idx_s, basis, mnb, bWc, bWd, xb);
  k3_ts         <<<grd, blk, 0, stream>>>(xb, bWts, xts);
  k4_st_combine <<<grd, blk, 0, stream>>>(xb, bWst, xts, idxsw, (float*)d_out);
}
